// UGRNNNet_25357486916090
// MI455X (gfx1250) — compile-verified, matrix operand order checked by reading, not run on hardware
//
#include <hip/hip_runtime.h>
#include <hip/hip_bf16.h>
#include <stdint.h>

// ---------------------------------------------------------------------------
// UGRNN on MI455X (gfx1250): bf16 WMMA, x-projection hoisted out of the scan,
// TDM (tensor_load_to_lds) double-buffered weight staging in the big GEMM,
// VMEM->LDS staging in the latency-critical step kernel.
// ---------------------------------------------------------------------------
#define B_   64
#define S_   512
#define E_   512
#define H_   512
#define C_   128
#define LDW_ 1024   // row stride of Wz / Wh ( = E_+H_ )
#define LPAD 40     // LDS row pitch in bf16 (80B = 20 banks -> conflict-free b128)

typedef __bf16 bf16_t;
typedef __attribute__((ext_vector_type(16))) __bf16 v16bf;
typedef __attribute__((ext_vector_type(8)))  __bf16 v8bf;
typedef __attribute__((ext_vector_type(8)))  float  v8f;
typedef unsigned int u32;
typedef __attribute__((ext_vector_type(4))) unsigned int u32x4;
typedef __attribute__((ext_vector_type(8))) int         i32x8;
typedef __attribute__((ext_vector_type(4))) int         i32x4;

#if defined(__has_builtin)
#if __has_builtin(__builtin_amdgcn_tensor_load_to_lds) && \
    __has_builtin(__builtin_amdgcn_s_wait_tensorcnt)
#define HAVE_TDM 1
#endif
#endif

__device__ __forceinline__ bf16_t f2bf(float f) { return (bf16_t)f; }  // hw cvt

// Wave32 16x16x32 bf16 fragment for one lane from a row-major matrix row.
// A layout (ISA 7.12.2): lanes 0-15 hold K={kb+0..7, kb+16..23}, lanes 16-31
// hold K={kb+8..15, kb+24..31}.  B symmetric with lane = N.
__device__ __forceinline__ v16bf frag_row(const bf16_t* row, int kb, int laneHi) {
  const v8bf a = *(const v8bf*)(row + kb + (laneHi ? 8 : 0));
  const v8bf b = *(const v8bf*)(row + kb + (laneHi ? 24 : 16));
  return __builtin_shufflevector(a, b, 0, 1, 2, 3, 4, 5, 6, 7,
                                       8, 9, 10, 11, 12, 13, 14, 15);
}

__device__ __forceinline__ v8f wmma_bf16(v16bf a, v16bf b, v8f c) {
  return __builtin_amdgcn_wmma_f32_16x16x32_bf16(
      false, a, false, b, (short)0, c, false, false);
}

__device__ __forceinline__ void cvt8(const float* __restrict__ s,
                                     bf16_t* __restrict__ d) {
  float4 f0 = ((const float4*)s)[0];
  float4 f1 = ((const float4*)s)[1];
  v8bf o;
  o[0] = f2bf(f0.x); o[1] = f2bf(f0.y); o[2] = f2bf(f0.z); o[3] = f2bf(f0.w);
  o[4] = f2bf(f1.x); o[5] = f2bf(f1.y); o[6] = f2bf(f1.z); o[7] = f2bf(f1.w);
  *(v8bf*)d = o;
}

#ifdef HAVE_TDM
// TDM 2-D tile copy: 64 rows x 32 bf16 from row-major W (pitch LDW_) into LDS
// with pitch LPAD (pad_interval=3: every 16 DWORDs, pad_amount=3: +4 DWORDs).
// D# layout per CDNA5 ISA.
__device__ __forceinline__ void tdm_load_tile(u32 lds_off, const bf16_t* gsrc) {
  uint64_t ga = (uint64_t)(uintptr_t)gsrc;
  u32x4 g0;
  g0[0] = 1u;                                        // count=1, user mode
  g0[1] = lds_off;                                   // lds_addr (bytes)
  g0[2] = (u32)ga;                                   // global_addr[31:0]
  g0[3] = (u32)((ga >> 32) & 0x1FFFFFFu) | (2u << 30);  // addr[56:32] | type=2
  i32x8 g1;
  g1[0] = (1 << 16) | (1 << 20) | (3 << 22) | (3 << 25); // 2B elems, pad en,
                                                         // interval=16dw, amt=4dw
  g1[1] = (32 << 16);          // tensor_dim0 = 32 (bits 79:48, low half)
  g1[2] = (64 << 16);          // tensor_dim0 hi = 0 | tensor_dim1 = 64 (lo)
  g1[3] = (32 << 16);          // tensor_dim1 hi = 0 | tile_dim0 = 32
  g1[4] = 64;                  // tile_dim1 = 64, tile_dim2 = 0
  g1[5] = LDW_;                // tensor_dim0_stride = 1024 elements
  g1[6] = 0;                   // stride hi, dim1_stride lo (unused for 2D)
  g1[7] = 0;
  i32x4 z4 = {0, 0, 0, 0};
#if defined(__clang_major__) && __clang_major__ >= 23
  i32x8 z8 = {0, 0, 0, 0, 0, 0, 0, 0};
  __builtin_amdgcn_tensor_load_to_lds(g0, g1, z4, z4, z8, 0);
#else
  __builtin_amdgcn_tensor_load_to_lds(g0, g1, z4, z4, 0);
#endif
}
#endif

// ---------------------------------------------------------------------------
// 0a) one-time f32 -> bf16 weight conversion (8 elements / thread)
// ---------------------------------------------------------------------------
__global__ void cvt_weights_kernel(const float* __restrict__ Wz,
                                   const float* __restrict__ Wh,
                                   const float* __restrict__ fcW,
                                   bf16_t* __restrict__ Wz_bf,
                                   bf16_t* __restrict__ Wh_bf,
                                   bf16_t* __restrict__ fc_bf) {
  const int nW8 = (H_ * LDW_) / 8;
  const int nF8 = (C_ * H_) / 8;
  int t = blockIdx.x * blockDim.x + threadIdx.x;
  if (t < nW8)                 cvt8(Wz  + (size_t)t * 8,             Wz_bf + (size_t)t * 8);
  else if (t < 2 * nW8)        cvt8(Wh  + (size_t)(t - nW8) * 8,     Wh_bf + (size_t)(t - nW8) * 8);
  else if (t < 2 * nW8 + nF8)  cvt8(fcW + (size_t)(t - 2 * nW8) * 8, fc_bf + (size_t)(t - 2 * nW8) * 8);
}

// ---------------------------------------------------------------------------
// 0b) gather + convert embeddings once: Ebf[r, :] = bf16(emb[x[r]]), r = s*B+b
// ---------------------------------------------------------------------------
__global__ void gather_emb_kernel(const int* __restrict__ x,
                                  const float* __restrict__ emb,
                                  bf16_t* __restrict__ Ebf) {
  int t = blockIdx.x * blockDim.x + threadIdx.x;
  int r = t >> 6;
  int k = (t & 63) * 8;
  int s = r >> 6, b = r & 63;
  int tok = x[b * S_ + s];
  cvt8(emb + (size_t)tok * E_ + k, Ebf + (size_t)r * E_ + k);
}

__global__ void init_h_kernel(float* __restrict__ h_f, bf16_t* __restrict__ h_bf) {
  int i = blockIdx.x * blockDim.x + threadIdx.x;
  if (i < B_ * H_) { h_f[i] = 0.0f; h_bf[i] = f2bf(0.0f); }
}

// ---------------------------------------------------------------------------
// 1) x-projection GEMM: G[r, n] = Ebf[r,:] . W{z|h}[n%512, 0:E] + bias
//    wave tile = 16x64; weight panel staged in LDS (TDM double-buffered)
// ---------------------------------------------------------------------------
__global__ void __launch_bounds__(128)
xproj_kernel(const bf16_t* __restrict__ Ebf,
             const bf16_t* __restrict__ Wz_bf, const bf16_t* __restrict__ Wh_bf,
             const float* __restrict__ bz, const float* __restrict__ bh,
             float* __restrict__ G) {
  __shared__ bf16_t lb[2][64][LPAD];                // double-buffered B panel

  const int tid  = threadIdx.x;
  const int lane = tid & 31;
  const int wave = tid >> 5;
  const int hi   = lane >> 4;
  const int l15  = lane & 15;
  const int n0   = blockIdx.x * 64;                 // 16 groups over 1024 cols
  const int r0   = (blockIdx.y * 4 + wave) * 16;    // row tile over B*S

  const bf16_t* arow = Ebf + (size_t)(r0 + l15) * E_;
  // whole 64-col group lives in one matrix (n0 is 64-aligned, H_ % 64 == 0)
  const bf16_t* wbase = (n0 < H_) ? (Wz_bf + (size_t)n0 * LDW_)
                                  : (Wh_bf + (size_t)(n0 - H_) * LDW_);

  v8f acc[4] = {};

#ifdef HAVE_TDM
  const int wave0 = (__builtin_amdgcn_readfirstlane(tid >> 5) == 0);
  const u32 lds0 = (u32)(uintptr_t)&lb[0][0][0];
  const u32 lds1 = (u32)(uintptr_t)&lb[1][0][0];

  if (wave0) {                                      // prologue: chunk 0 -> buf0
    tdm_load_tile(lds0, wbase);
    __builtin_amdgcn_s_wait_tensorcnt(0);
  }
  __syncthreads();

  for (int kb = 0; kb < E_; kb += 64) {
    if (wave0) tdm_load_tile(lds1, wbase + kb + 32);     // DMA next chunk
    {
      v16bf a = frag_row(arow, kb, hi);
#pragma unroll
      for (int t = 0; t < 4; ++t)
        acc[t] = wmma_bf16(a, frag_row(&lb[0][t * 16 + l15][0], 0, hi), acc[t]);
    }
    if (wave0) __builtin_amdgcn_s_wait_tensorcnt(0);
    __syncthreads();

    if (wave0 && (kb + 64 < E_)) tdm_load_tile(lds0, wbase + kb + 64);
    {
      v16bf a = frag_row(arow, kb + 32, hi);
#pragma unroll
      for (int t = 0; t < 4; ++t)
        acc[t] = wmma_bf16(a, frag_row(&lb[1][t * 16 + l15][0], 0, hi), acc[t]);
    }
    if (wave0) __builtin_amdgcn_s_wait_tensorcnt(0);
    __syncthreads();
  }
#else
  // fallback: cooperative VMEM -> LDS staging into buffer 0
  const int ri4 = tid >> 2;
  const int q   = tid & 3;
  const bf16_t* wsrc[2];
  bf16_t* ldst[2];
#pragma unroll
  for (int p = 0; p < 2; ++p) {
    int r = ri4 + p * 32;
    wsrc[p] = wbase + (size_t)r * LDW_ + q * 8;
    ldst[p] = &lb[0][r][q * 8];
  }
  for (int kb = 0; kb < E_; kb += 32) {
    *(v8bf*)ldst[0] = *(const v8bf*)(wsrc[0] + kb);
    *(v8bf*)ldst[1] = *(const v8bf*)(wsrc[1] + kb);
    __syncthreads();
    v16bf a = frag_row(arow, kb, hi);
#pragma unroll
    for (int t = 0; t < 4; ++t)
      acc[t] = wmma_bf16(a, frag_row(&lb[0][t * 16 + l15][0], 0, hi), acc[t]);
    __syncthreads();
  }
#endif

#pragma unroll
  for (int t = 0; t < 4; ++t) {
    int   n    = n0 + t * 16 + l15;
    float bias = (n < H_) ? bz[n] : bh[n - H_];
#pragma unroll
    for (int j = 0; j < 8; ++j) {
      int m = j + hi * 8;                            // C/D layout: VGPR j -> M
      G[(size_t)(r0 + m) * (2 * H_) + n] = acc[t][j] + bias;
    }
  }
}

// ---------------------------------------------------------------------------
// 2) one recurrent step: both gates' h-projections + fused gate epilogue,
//    weight tiles staged in LDS (shared by the 4 waves)
// ---------------------------------------------------------------------------
__global__ void __launch_bounds__(128)
step_kernel(const bf16_t* __restrict__ hprev_bf, const float* __restrict__ hprev_f,
            const bf16_t* __restrict__ Wz_bf, const bf16_t* __restrict__ Wh_bf,
            const float* __restrict__ G, int s,
            bf16_t* __restrict__ hcur_bf, float* __restrict__ hcur_f) {
  __shared__ bf16_t lbz[16][LPAD];
  __shared__ bf16_t lbh[16][LPAD];

  const int tid  = threadIdx.x;
  const int lane = tid & 31;
  const int wave = tid >> 5;
  const int hi   = lane >> 4;
  const int l15  = lane & 15;
  const int n0   = blockIdx.x * 16;   // 0..511
  const int r0   = wave * 16;         // 4 waves cover B=64 rows

  const bf16_t* arow = hprev_bf + (size_t)(r0 + l15) * H_;

  const int g  = tid >> 6;            // 0 = z, 1 = h
  const int ri = (tid >> 2) & 15;
  const int q  = tid & 3;
  const bf16_t* wsrc = ((g ? Wh_bf : Wz_bf) + (size_t)(n0 + ri) * LDW_ + E_) + q * 8;
  bf16_t* ldst = (g ? &lbh[ri][0] : &lbz[ri][0]) + q * 8;

  v8f accZ = {}, accH = {};
  for (int kb = 0; kb < H_; kb += 32) {
    *(v8bf*)ldst = *(const v8bf*)(wsrc + kb);
    __syncthreads();
    v16bf a   = frag_row(arow, kb, hi);
    v16bf bzt = frag_row(&lbz[l15][0], 0, hi);
    v16bf bht = frag_row(&lbh[l15][0], 0, hi);
    accZ = wmma_bf16(a, bzt, accZ);
    accH = wmma_bf16(a, bht, accH);
    __syncthreads();
  }

  const int    n    = n0 + l15;
  const float* Grow = G + (size_t)s * B_ * (2 * H_);
#pragma unroll
  for (int j = 0; j < 8; ++j) {
    int   row = r0 + j + hi * 8;
    float gz  = Grow[(size_t)row * (2 * H_) + n];
    float gh  = Grow[(size_t)row * (2 * H_) + H_ + n];
    float z   = 1.0f / (1.0f + __expf(-(accZ[j] + gz)));
    float ht  = tanhf(accH[j] + gh);
    float hp  = hprev_f[(size_t)row * H_ + n];
    float hn  = z * hp + (1.0f - z) * ht;
    hcur_f[(size_t)row * H_ + n]  = hn;
    hcur_bf[(size_t)row * H_ + n] = f2bf(hn);
  }
}

// ---------------------------------------------------------------------------
// 3) final FC: out[64,128] = h_final . fcW^T + fcb
// ---------------------------------------------------------------------------
__global__ void __launch_bounds__(128)
fc_kernel(const bf16_t* __restrict__ h_bf, const bf16_t* __restrict__ fc_bf,
          const float* __restrict__ fcb, float* __restrict__ out) {
  const int lane = threadIdx.x & 31;
  const int wave = threadIdx.x >> 5;
  const int hi   = lane >> 4;
  const int l15  = lane & 15;
  const int n0   = blockIdx.x * 16;   // 0..127
  const int r0   = wave * 16;

  const bf16_t* arow = h_bf  + (size_t)(r0 + l15) * H_;
  const bf16_t* brow = fc_bf + (size_t)(n0 + l15) * H_;

  v8f acc = {};
#pragma unroll 4
  for (int kb = 0; kb < H_; kb += 32) {
    v16bf a  = frag_row(arow, kb, hi);
    v16bf bb = frag_row(brow, kb, hi);
    acc = wmma_bf16(a, bb, acc);
  }

  const int n = n0 + l15;
#pragma unroll
  for (int j = 0; j < 8; ++j) {
    int row = r0 + j + hi * 8;
    out[(size_t)row * C_ + n] = acc[j] + fcb[n];
  }
}

// ---------------------------------------------------------------------------
extern "C" void kernel_launch(void* const* d_in, const int* in_sizes, int n_in,
                              void* d_out, int out_size, void* d_ws, size_t ws_size,
                              hipStream_t stream) {
  (void)in_sizes; (void)n_in; (void)out_size; (void)ws_size;

  const int*   x   = (const int*)  d_in[0];
  const float* emb = (const float*)d_in[1];
  const float* Wz  = (const float*)d_in[2];
  const float* bz  = (const float*)d_in[3];
  const float* Wh  = (const float*)d_in[4];
  const float* bh  = (const float*)d_in[5];
  const float* fcW = (const float*)d_in[6];
  const float* fcb = (const float*)d_in[7];

  char* ws = (char*)d_ws;
  size_t o = 0;
  bf16_t* Wz_bf = (bf16_t*)(ws + o); o += (size_t)H_ * LDW_ * sizeof(bf16_t);
  bf16_t* Wh_bf = (bf16_t*)(ws + o); o += (size_t)H_ * LDW_ * sizeof(bf16_t);
  bf16_t* fc_bf = (bf16_t*)(ws + o); o += (size_t)C_ * H_  * sizeof(bf16_t);
  float*  hA_f  = (float*) (ws + o); o += (size_t)B_ * H_  * sizeof(float);
  float*  hB_f  = (float*) (ws + o); o += (size_t)B_ * H_  * sizeof(float);
  bf16_t* hA_bf = (bf16_t*)(ws + o); o += (size_t)B_ * H_  * sizeof(bf16_t);
  bf16_t* hB_bf = (bf16_t*)(ws + o); o += (size_t)B_ * H_  * sizeof(bf16_t);
  bf16_t* Ebf   = (bf16_t*)(ws + o); o += (size_t)B_ * S_ * E_ * sizeof(bf16_t); // 33.5 MB
  float*  G     = (float*) (ws + o);  // [B*S, 2H] f32, ~134 MB

  {
    int chunks = 2 * (H_ * LDW_) / 8 + (C_ * H_) / 8;
    cvt_weights_kernel<<<(chunks + 255) / 256, 256, 0, stream>>>(Wz, Wh, fcW,
                                                                 Wz_bf, Wh_bf, fc_bf);
  }
  gather_emb_kernel<<<(B_ * S_ * 64) / 256, 256, 0, stream>>>(x, emb, Ebf);
  init_h_kernel<<<(B_ * H_ + 255) / 256, 256, 0, stream>>>(hA_f, hA_bf);

  // 1) hoisted x-projection GEMM: [32768 x 512] x [512 x 1024]
  xproj_kernel<<<dim3((2 * H_) / 64, (B_ * S_) / 64), 128, 0, stream>>>(
      Ebf, Wz_bf, Wh_bf, bz, bh, G);

  // 2) sequential recurrence, ping-pong h buffers
  for (int s = 0; s < S_; ++s) {
    const bf16_t* hp_b = (s & 1) ? hB_bf : hA_bf;
    const float*  hp_f = (s & 1) ? hB_f  : hA_f;
    bf16_t*       hc_b = (s & 1) ? hA_bf : hB_bf;
    float*        hc_f = (s & 1) ? hA_f  : hB_f;
    step_kernel<<<H_ / 16, 128, 0, stream>>>(hp_b, hp_f, Wz_bf, Wh_bf, G, s, hc_b, hc_f);
  }
  // after 512 steps (even count), final h lives in the A buffers

  // 3) final FC
  fc_kernel<<<C_ / 16, 128, 0, stream>>>(hA_bf, fc_bf, fcb, (float*)d_out);
}
